// DSNN_49185965474303
// MI455X (gfx1250) — matrix operand order repaired and checked by reading, not compiled, not run on hardware
//
#include <hip/hip_runtime.h>

typedef __attribute__((ext_vector_type(16))) _Float16 v16h;
typedef __attribute__((ext_vector_type(8)))  _Float16 v8h;
typedef __attribute__((ext_vector_type(8)))  float    v8f;

#define ALPHA   0.95f
#define BETA    0.9f
#define NSTEPS  99          // SIM_TIME - 1 update steps
#define H0      256
#define H1      512
#define H2      512
#define HOUT    4
#define BATCH   512
#define LSTR    520         // padded LDS row stride (halves): 1040B, 16B-aligned, conflict-free

union ABFrag { v16h v; v8h h[2]; };

__device__ __forceinline__ v16h make_frag(v8h lo, v8h hi) {
  ABFrag u; u.h[0] = lo; u.h[1] = hi; return u.v;
}

__device__ __forceinline__ v8f wmma_f16(v16h a, v16h b, v8f c) {
  return __builtin_amdgcn_wmma_f32_16x16x32_f16(false, a, false, b, (short)0, c, false, false);
}

// ---------------------------------------------------------------------------
// Prep: fp32 -> f16, weights transposed so B-fragments are contiguous loads.
//   XH  [BATCH][H0]  = X
//   W0T [H1][H0]     = W0^T
//   W1T [H2][H1]     = W1^T
//   W2T [16][H2]     = W2^T zero-padded from 4 to 16 output columns
// ---------------------------------------------------------------------------
__global__ void snn_prep(const float* __restrict__ X,  const float* __restrict__ W0,
                         const float* __restrict__ W1, const float* __restrict__ W2,
                         _Float16* __restrict__ XH,  _Float16* __restrict__ W0T,
                         _Float16* __restrict__ W1T, _Float16* __restrict__ W2T) {
  const int r0 = BATCH * H0;          // XH
  const int r1 = r0 + H1 * H0;        // W0T
  const int r2 = r1 + H2 * H1;        // W1T
  const int r3 = r2 + 16 * H2;        // W2T
  for (int i = blockIdx.x * blockDim.x + threadIdx.x; i < r3;
       i += gridDim.x * blockDim.x) {
    if (i < r0) {
      XH[i] = (_Float16)X[i];
    } else if (i < r1) {
      int j = i - r0, n = j / H0, k = j % H0;
      W0T[j] = (_Float16)W0[k * H1 + n];
    } else if (i < r2) {
      int j = i - r1, n = j / H1, k = j % H1;
      W1T[j] = (_Float16)W1[k * H2 + n];
    } else {
      int j = i - r2, n = j / H2, k = j % H2;
      W2T[j] = (n < HOUT) ? (_Float16)W2[k * HOUT + n] : (_Float16)0.0f;
    }
  }
}

// ---------------------------------------------------------------------------
// Main: one workgroup = 16 batch rows, 8 waves, all 99 timesteps locally.
// ---------------------------------------------------------------------------
__global__ __launch_bounds__(256, 1)
void snn_main(const _Float16* __restrict__ XH,  const _Float16* __restrict__ W0T,
              const _Float16* __restrict__ W1T, const _Float16* __restrict__ W2T,
              float* __restrict__ out) {
  __shared__ __align__(16) _Float16 s0L[16 * LSTR];   // layer-0 spikes (A layout)
  __shared__ __align__(16) _Float16 s1L[16 * LSTR];   // layer-1 spikes (A layout)
  __shared__ float acc2[16 * 16];                      // layer-2 K-split reduction

  const int tid   = threadIdx.x;
  const int lane  = tid & 31;
  const int wave  = tid >> 5;        // 0..7
  const int nlo   = lane & 15;       // A: M row / B: N col / C: N col
  const int hi    = lane >> 4;       // lane half selects K sub-block / M+8
  const int koffA = hi * 8;          // A-frag K offset within 32-K step
  const int kbB   = hi * 8;          // B-frag K offset within 32-K step
  const int b0    = blockIdx.x * 16; // batch tile base

  // -------- I0 = XH[b0:b0+16,:] @ W0  (once; 4 hidden1 tiles per wave) -----
  v8f I0[4], mem0[4], syn1[4], mem1[4];
  {
    v8f acc[4] = {{}, {}, {}, {}};
    const _Float16* arow = XH + (size_t)(b0 + nlo) * H0;
    for (int ks = 0; ks < H0; ks += 32) {
      v16h a = make_frag(*(const v8h*)(arow + ks + koffA),
                         *(const v8h*)(arow + ks + 16 + koffA));
#pragma unroll
      for (int t = 0; t < 4; t++) {
        const int col0 = (wave * 4 + t) * 16;
        const _Float16* bp = W0T + (size_t)(col0 + nlo) * H0 + ks + kbB;
        v16h b = make_frag(*(const v8h*)bp, *(const v8h*)(bp + 16));
        acc[t] = wmma_f16(a, b, acc[t]);
      }
    }
#pragma unroll
    for (int t = 0; t < 4; t++) {
      I0[t] = acc[t];
      mem0[t] = (v8f){}; syn1[t] = (v8f){}; mem1[t] = (v8f){};
    }
  }

  v8f syn2 = {}, mem2 = {};
  if (wave == 0) {
#pragma unroll
    for (int r = 0; r < 8; r++) acc2[(r + hi * 8) * 16 + nlo] = 0.0f;
  }
  __syncthreads();

  // ---------------------------- time loop ---------------------------------
  for (int t = 0; t < NSTEPS; t++) {
    // Phase A: layer-0 integrate / spike / reset -> s0 to LDS (f16, A layout)
#pragma unroll
    for (int tl = 0; tl < 4; tl++) {
      const int col0 = (wave * 4 + tl) * 16;
#pragma unroll
      for (int r = 0; r < 8; r++) {
        float nm   = BETA * mem0[tl][r] + I0[tl][r];
        bool  spk  = (nm - 1.0f) > 0.0f;
        s0L[(r + hi * 8) * LSTR + col0 + nlo] = spk ? (_Float16)1.0f : (_Float16)0.0f;
        mem0[tl][r] = spk ? 0.0f : nm;
      }
    }
    __syncthreads();

    // Phase B: layer-1  ns1 = ALPHA*syn1 + s0 @ W1   (WMMA, K=512)
    {
      v8f acc[4] = {{}, {}, {}, {}};
      for (int ks = 0; ks < H1; ks += 32) {
        const _Float16* ap = &s0L[nlo * LSTR + ks + koffA];
        v16h a = make_frag(*(const v8h*)ap, *(const v8h*)(ap + 16));
#pragma unroll
        for (int tl = 0; tl < 4; tl++) {
          const int col0 = (wave * 4 + tl) * 16;
          const _Float16* bp = W1T + (size_t)(col0 + nlo) * H1 + ks + kbB;
          v16h b = make_frag(*(const v8h*)bp, *(const v8h*)(bp + 16));
          acc[tl] = wmma_f16(a, b, acc[tl]);
        }
      }
#pragma unroll
      for (int tl = 0; tl < 4; tl++) {
        const int col0 = (wave * 4 + tl) * 16;
#pragma unroll
        for (int r = 0; r < 8; r++) {
          float ns  = ALPHA * syn1[tl][r] + acc[tl][r];
          float nm  = BETA  * mem1[tl][r] + ns;
          bool  spk = (nm - 1.0f) > 0.0f;
          s1L[(r + hi * 8) * LSTR + col0 + nlo] = spk ? (_Float16)1.0f : (_Float16)0.0f;
          syn1[tl][r] = ns;
          mem1[tl][r] = spk ? 0.0f : nm;
        }
      }
    }
    __syncthreads();

    // Phase C: layer-2 partials, K slice [wave*64, wave*64+64) -> LDS atomics
    {
      v8f c2 = {};
#pragma unroll
      for (int kk = 0; kk < 2; kk++) {
        const int ks = wave * 64 + kk * 32;
        const _Float16* ap = &s1L[nlo * LSTR + ks + koffA];
        v16h a = make_frag(*(const v8h*)ap, *(const v8h*)(ap + 16));
        const _Float16* bp = W2T + (size_t)nlo * H2 + ks + kbB;
        v16h b = make_frag(*(const v8h*)bp, *(const v8h*)(bp + 16));
        c2 = wmma_f16(a, b, c2);
      }
#pragma unroll
      for (int r = 0; r < 8; r++)
        atomicAdd(&acc2[(r + hi * 8) * 16 + nlo], c2[r]);
    }
    __syncthreads();

    // Phase D: wave 0 integrates the output leaky integrator, re-zeros acc2
    if (wave == 0) {
#pragma unroll
      for (int r = 0; r < 8; r++) {
        const int idx = (r + hi * 8) * 16 + nlo;
        float ns = ALPHA * syn2[r] + acc2[idx];
        float nm = BETA  * mem2[r] + ns;
        syn2[r] = ns;
        mem2[r] = nm;
        acc2[idx] = 0.0f;
      }
    }
    __syncthreads();
  }

  // final membrane of output layer -> d_out [512][4] fp32
  if (wave == 0 && nlo < HOUT) {
#pragma unroll
    for (int r = 0; r < 8; r++)
      out[(size_t)(b0 + r + hi * 8) * HOUT + nlo] = mem2[r];
  }
}

// ---------------------------------------------------------------------------
extern "C" void kernel_launch(void* const* d_in, const int* in_sizes, int n_in,
                              void* d_out, int out_size, void* d_ws, size_t ws_size,
                              hipStream_t stream) {
  (void)in_sizes; (void)n_in; (void)out_size; (void)ws_size;
  const float* X  = (const float*)d_in[0];
  const float* W0 = (const float*)d_in[1];
  const float* W1 = (const float*)d_in[2];
  const float* W2 = (const float*)d_in[3];
  float* out = (float*)d_out;

  _Float16* ws  = (_Float16*)d_ws;
  _Float16* XH  = ws;                 // 131072 halves
  _Float16* W0T = ws + 131072;        // 131072 halves
  _Float16* W1T = ws + 262144;        // 262144 halves
  _Float16* W2T = ws + 524288;        //   8192 halves   (~1.04 MB total)

  snn_prep<<<512, 256, 0, stream>>>(X, W0, W1, W2, XH, W0T, W1T, W2T);
  snn_main<<<BATCH / 16, 256, 0, stream>>>(XH, W0T, W1T, W2T, out);
}
